// ZNCC_NN_1082331758901
// MI455X (gfx1250) — matrix operand order repaired and checked by reading, not compiled, not run on hardware
//
#include <hip/hip_runtime.h>
#include <hip/hip_bf16.h>
#include <stdint.h>

// ---------------------------------------------------------------------------
// ZNCC structured-light matcher, fused for MI455X (gfx1250, wave32, WMMA).
//   P=7, N_PAT=32, N_FEAT=224, H=256, W_IMG=512, W_PAT=1024, 2 residual layers
//   * native __bf16 data / f32 accumulate via v_wmma_f32_16x16x32_bf16
//   * weights pre-packed into WMMA B fragments; staged to LDS via TDM
//     (tensor_load_to_lds + s_wait_tensorcnt)
//   * A fragments hoisted to VGPRs; B fragments batch-loaded -> back-to-back
//     WMMA chains; fully parallel ZNCC normalize
// ---------------------------------------------------------------------------

#define NF      224
#define LDX     224          // LDS row stride (elements); 448 B, 16B aligned
#define ROWS_WG 128          // rows (pixels / patterns) per workgroup
#define KT_N    7            // 224 / 32 K-tiles
#define NT_MLP  14           // 224 / 16 N-tiles inside MLP
#define NT_OUT  64           // 1024 / 16 N-tiles in final correlation
#define WFRAG_MAT 50176      // 7*14*32*16 bf16 elements per packed matrix
#define SMEM_BYTES (3*ROWS_WG*LDX*2 + WFRAG_MAT*2)   // 172032 + 100352 = 272384

typedef __attribute__((ext_vector_type(16))) __bf16 v16bf;
typedef __attribute__((ext_vector_type(8)))  float  v8f;
typedef __attribute__((ext_vector_type(4)))  unsigned int u32x4;
typedef __attribute__((ext_vector_type(8)))  int i32x8;
typedef __attribute__((ext_vector_type(4)))  int i32x4;

union Frag16 { uint4 q[2]; v16bf v; };

#if defined(__has_builtin)
#if __has_builtin(__builtin_amdgcn_tensor_load_to_lds)
#define HAVE_TDM 1
#endif
#endif

// ---------------------------------------------------------------------------
// Stage one packed weight matrix (100,352 B) global -> LDS.
// TDM path: 1-row tensor descriptor (D#), one instruction per workgroup,
// issued by wave 0 only (EXEC ignored; TENSORcnt is per-wave, barrier after).
// ---------------------------------------------------------------------------
__device__ __forceinline__ void stage_wl(__bf16* Wl,
                                         const __bf16* __restrict__ src,
                                         int tid)
{
#if HAVE_TDM
    if ((tid >> 5) == 0) {
        unsigned int lds = (unsigned int)(uintptr_t)Wl;          // LDS byte addr
        unsigned long long ga = (unsigned long long)(uintptr_t)src;
        u32x4 g0;
        g0[0] = 1u;                                   // count=1 (valid user D#)
        g0[1] = lds;                                  // lds_addr  [63:32]
        g0[2] = (unsigned int)ga;                     // global_addr[31:0]
        g0[3] = (unsigned int)(ga >> 32) | (2u << 30);// global_addr[56:32]|type=2
        i32x8 g1;
        g1[0] = 0x10000;                              // data_size=1 -> 2 bytes
        g1[1] = (int)((unsigned)WFRAG_MAT << 16);     // tensor_dim0[15:0]
        g1[2] = 0x10000;                              // tensor_dim1 = 1
        g1[3] = (int)((unsigned)WFRAG_MAT << 16);     // tile_dim0 = 50176
        g1[4] = 1;                                    // tile_dim1 = 1
        g1[5] = WFRAG_MAT;                            // tensor_dim0_stride
        g1[6] = 0;
        g1[7] = 0;
        i32x4 g2 = {};
        i32x4 g3 = {};
        i32x8 g4 = {};
        __builtin_amdgcn_tensor_load_to_lds(g0, g1, g2, g3, g4, 0);
        __builtin_amdgcn_s_wait_tensorcnt(0);
    }
#else
    {
        const uint4* s = (const uint4*)src;
        uint4* d = (uint4*)Wl;
        for (int i = tid; i < WFRAG_MAT / 8; i += 256) d[i] = s[i];
    }
#endif
}

// ---------------------------------------------------------------------------
// Kernel 0: pack 8 weight matrices (f32, torch Linear layout [out][in]) into
// bf16 WMMA-B fragments:  wfrag[mat][kt][nt][lane][j],  B[k][n] = W[n][k].
// ---------------------------------------------------------------------------
__global__ void prep_wfrag(const float* __restrict__ cam_w0,
                           const float* __restrict__ cam_w1,
                           const float* __restrict__ proj_w0,
                           const float* __restrict__ proj_w1,
                           __bf16* __restrict__ wfrag)
{
    int idx = blockIdx.x * 256 + threadIdx.x;
    if (idx >= 8 * WFRAG_MAT) return;
    int mat = idx / WFRAG_MAT;
    int rem = idx - mat * WFRAG_MAT;
    int j    = rem & 15;
    int lane = (rem >> 4) & 31;
    int t    = rem >> 9;
    int nt   = t % NT_MLP;
    int kt   = t / NT_MLP;
    int hf   = lane >> 4;
    int n    = nt * 16 + (lane & 15);
    int k    = kt * 32 + ((j >> 3) << 4) + hf * 8 + (j & 7);
    int mm = mat & 3, l = mm >> 1, which = mm & 1;
    const float* W = (mat < 4) ? (which ? cam_w1 : cam_w0)
                               : (which ? proj_w1 : proj_w0);
    wfrag[idx] = (__bf16)W[((size_t)l * NF + n) * NF + k];
}

// ---------------------------------------------------------------------------
// Shared: 2 residual MLP layers (WMMA) + (x += x0) + per-row ZNCC normalize.
// Each wave owns 16 rows; all A/C accesses are row-local to the wave.
// ---------------------------------------------------------------------------
__device__ __forceinline__ void mlp_and_norm(
    __bf16* X0, __bf16* X, __bf16* Hs, __bf16* Wl,
    const __bf16* __restrict__ wfrag, int matBase,
    const float* __restrict__ b0, const float* __restrict__ b1, int tid)
{
    const int lane = tid & 31;
    const int row0 = (tid >> 5) * 16;
    const int l15  = lane & 15;
    const int hf   = lane >> 4;

    for (int l = 0; l < 2; ++l) {
        // ---- stage W0[l] fragments into LDS ----
        __syncthreads();
        stage_wl(Wl, wfrag + (size_t)(matBase + l*2 + 0) * WFRAG_MAT, tid);
        __syncthreads();
        // ---- H = relu(X @ W0^T + b0[l]) ----
        {
            Frag16 Afr[KT_N];
            #pragma unroll
            for (int kt = 0; kt < KT_N; ++kt) {
                const __bf16* ap = &X[(row0 + l15) * LDX + kt * 32 + hf * 8];
                Afr[kt].q[0] = *(const uint4*)ap;
                Afr[kt].q[1] = *(const uint4*)(ap + 16);
            }
            for (int nt = 0; nt < NT_MLP; ++nt) {
                float bias = b0[l * NF + nt * 16 + l15];
                v8f acc;
                #pragma unroll
                for (int r = 0; r < 8; ++r) acc[r] = bias;
                Frag16 Bfr[KT_N];
                #pragma unroll
                for (int kt = 0; kt < KT_N; ++kt) {
                    const uint4* bp = (const uint4*)&Wl[((kt * NT_MLP + nt) * 32 + lane) * 16];
                    Bfr[kt].q[0] = bp[0]; Bfr[kt].q[1] = bp[1];
                }
                #pragma unroll
                for (int kt = 0; kt < KT_N; ++kt)
                    acc = __builtin_amdgcn_wmma_f32_16x16x32_bf16(
                              false, Afr[kt].v, false, Bfr[kt].v, (short)0, acc, false, false);
                #pragma unroll
                for (int r = 0; r < 8; ++r)
                    Hs[(row0 + r + hf * 8) * LDX + nt * 16 + l15] =
                        (__bf16)(acc[r] > 0.f ? acc[r] : 0.f);
            }
        }
        // ---- stage W1[l] fragments into LDS ----
        __syncthreads();
        stage_wl(Wl, wfrag + (size_t)(matBase + l*2 + 1) * WFRAG_MAT, tid);
        __syncthreads();
        // ---- X = X + H @ W1^T + b1[l]  (residual via C-operand init) ----
        {
            Frag16 Afr[KT_N];
            #pragma unroll
            for (int kt = 0; kt < KT_N; ++kt) {
                const __bf16* ap = &Hs[(row0 + l15) * LDX + kt * 32 + hf * 8];
                Afr[kt].q[0] = *(const uint4*)ap;
                Afr[kt].q[1] = *(const uint4*)(ap + 16);
            }
            for (int nt = 0; nt < NT_MLP; ++nt) {
                float bias = b1[l * NF + nt * 16 + l15];
                v8f acc;
                #pragma unroll
                for (int r = 0; r < 8; ++r)
                    acc[r] = bias + (float)X[(row0 + r + hf * 8) * LDX + nt * 16 + l15];
                Frag16 Bfr[KT_N];
                #pragma unroll
                for (int kt = 0; kt < KT_N; ++kt) {
                    const uint4* bp = (const uint4*)&Wl[((kt * NT_MLP + nt) * 32 + lane) * 16];
                    Bfr[kt].q[0] = bp[0]; Bfr[kt].q[1] = bp[1];
                }
                #pragma unroll
                for (int kt = 0; kt < KT_N; ++kt)
                    acc = __builtin_amdgcn_wmma_f32_16x16x32_bf16(
                              false, Afr[kt].v, false, Bfr[kt].v, (short)0, acc, false, false);
                #pragma unroll
                for (int r = 0; r < 8; ++r)
                    X[(row0 + r + hf * 8) * LDX + nt * 16 + l15] = (__bf16)acc[r];
            }
        }
    }
    __syncthreads();
    // ---- a = X + X0 ; zero-mean unit-norm per row (ZNCC) ----
    // all 256 threads: 2 threads per row, partial sums combined via f32
    // scratch carved from the (now free) Hs buffer.
    {
        float* scr = (float*)Hs;
        const int r  = tid & 127;
        const int ph = tid >> 7;
        const int k0 = ph * (NF / 2);
        float sum = 0.f, sq = 0.f;
        for (int k = k0; k < k0 + NF / 2; ++k) {
            float a = (float)X[r * LDX + k] + (float)X0[r * LDX + k];
            sum += a; sq += a * a;
            X[r * LDX + k] = (__bf16)a;
        }
        scr[(r * 2 + ph) * 2 + 0] = sum;
        scr[(r * 2 + ph) * 2 + 1] = sq;
        __syncthreads();
        float s = scr[(r * 2 + 0) * 2 + 0] + scr[(r * 2 + 1) * 2 + 0];
        float q = scr[(r * 2 + 0) * 2 + 1] + scr[(r * 2 + 1) * 2 + 1];
        float mean = s * (1.0f / NF);
        float var  = q - (float)NF * mean * mean;
        float rn   = rsqrtf(var > 1e-20f ? var : 1e-20f);
        for (int k = k0; k < k0 + NF / 2; ++k)
            X[r * LDX + k] = (__bf16)(((float)X[r * LDX + k] - mean) * rn);
    }
    __syncthreads();
}

// ---------------------------------------------------------------------------
// Kernel 1: projector side. 1024 patterns -> expand, MLP, normalize, scatter
// into fragment-packed B buffer bfrag[kt][nt 0..63][lane][16].
// ---------------------------------------------------------------------------
__global__ void proj_kernel(const float* __restrict__ proj_code,
                            const __bf16* __restrict__ wfrag,
                            const float* __restrict__ b0,
                            const float* __restrict__ b1,
                            __bf16* __restrict__ bfrag)
{
    extern __shared__ char smem[];
    __bf16* X0 = (__bf16*)smem;
    __bf16* X  = (__bf16*)(smem + 1 * ROWS_WG * LDX * 2);
    __bf16* Hs = (__bf16*)(smem + 2 * ROWS_WG * LDX * 2);
    __bf16* Wl = (__bf16*)(smem + 3 * ROWS_WG * LDX * 2);
    const int tid = threadIdx.x;
    const int M0  = blockIdx.x * ROWS_WG;

    // expand p_proj^T : row m, feature f = i*32+c -> proj_code[c][clip(m+i-3)]
    for (int idx = tid; idx < ROWS_WG * NF; idx += 256) {
        int r = idx / NF, f = idx - r * NF;
        int i = f >> 5, c = f & 31;
        int mc = M0 + r + i - 3;
        mc = mc < 0 ? 0 : (mc > 1023 ? 1023 : mc);
        __bf16 us = (__bf16)proj_code[(c << 10) + mc];
        X0[idx] = us; X[idx] = us;
    }

    mlp_and_norm(X0, X, Hs, Wl, wfrag, /*matBase=*/4, b0, b1, tid);

    // scatter normalized rows into packed B fragments
    for (int idx = tid; idx < ROWS_WG * NF; idx += 256) {
        int r = idx / NF, k = idx - r * NF;
        int m  = M0 + r;
        int kt = k >> 5, kr = k & 31;
        int bh = (kr >> 3) & 1, j = ((kr >> 4) << 3) | (kr & 7);
        int nt = m >> 4;
        int ln = bh * 16 + (m & 15);
        bfrag[(size_t)((kt * NT_OUT + nt) * 32 + ln) * 16 + j] = X[idx];
    }
}

// ---------------------------------------------------------------------------
// Kernel 2: camera side, fully fused. 128 pixels per WG (8 waves x 16 rows).
// expand -> MLP -> normalize -> 64 n-tiles of (16 x 1024 x 224) WMMA GEMM.
// ---------------------------------------------------------------------------
__global__ void cam_kernel(const float* __restrict__ cam_code,
                           const __bf16* __restrict__ wfrag,
                           const float* __restrict__ b0,
                           const float* __restrict__ b1,
                           const __bf16* __restrict__ bfrag,
                           float* __restrict__ out)
{
    extern __shared__ char smem[];
    __bf16* X0 = (__bf16*)smem;
    __bf16* X  = (__bf16*)(smem + 1 * ROWS_WG * LDX * 2);
    __bf16* Hs = (__bf16*)(smem + 2 * ROWS_WG * LDX * 2);
    __bf16* Wl = (__bf16*)(smem + 3 * ROWS_WG * LDX * 2);
    const int tid = threadIdx.x;
    const int R0  = blockIdx.x * ROWS_WG;

    // expand p_cam : pixel (h,w), feature f = i*32+c -> cam_code[h][clip(w+i-3)][c]
    for (int idx = tid; idx < ROWS_WG * NF; idx += 256) {
        int r = idx / NF, f = idx - r * NF;
        int i = f >> 5, c = f & 31;
        int pix = R0 + r;
        int h = pix >> 9, w = pix & 511;
        int wc = w + i - 3;
        wc = wc < 0 ? 0 : (wc > 511 ? 511 : wc);
        __bf16 us = (__bf16)cam_code[((size_t)((h << 9) + wc)) * 32 + c];
        X0[idx] = us; X[idx] = us;
    }

    mlp_and_norm(X0, X, Hs, Wl, wfrag, /*matBase=*/0, b0, b1, tid);

    // final ZNCC correlation: all 7 A fragments resident in VGPRs
    const int lane = tid & 31;
    const int row0 = (tid >> 5) * 16;
    const int l15  = lane & 15;
    const int hf   = lane >> 4;

    Frag16 Af[KT_N];
    #pragma unroll
    for (int kt = 0; kt < KT_N; ++kt) {
        const __bf16* ap = &X[(row0 + l15) * LDX + kt * 32 + hf * 8];
        Af[kt].q[0] = *(const uint4*)ap;
        Af[kt].q[1] = *(const uint4*)(ap + 16);
    }

    for (int nt = 0; nt < NT_OUT; ++nt) {
        // prefetch next n-tile's B fragments into L2/L0 while we compute
        if (nt + 1 < NT_OUT)
            __builtin_prefetch(&bfrag[(size_t)((nt + 1) * 32 + lane) * 16], 0, 3);

        Frag16 Bfr[KT_N];
        #pragma unroll
        for (int kt = 0; kt < KT_N; ++kt) {
            const uint4* bp = (const uint4*)&bfrag[(size_t)((kt * NT_OUT + nt) * 32 + lane) * 16];
            Bfr[kt].q[0] = bp[0]; Bfr[kt].q[1] = bp[1];
        }
        v8f acc;
        #pragma unroll
        for (int r = 0; r < 8; ++r) acc[r] = 0.f;
        #pragma unroll
        for (int kt = 0; kt < KT_N; ++kt)
            acc = __builtin_amdgcn_wmma_f32_16x16x32_bf16(
                      false, Af[kt].v, false, Bfr[kt].v, (short)0, acc, false, false);
        #pragma unroll
        for (int r = 0; r < 8; ++r)
            out[(size_t)(R0 + row0 + r + hf * 8) * 1024 + nt * 16 + l15] = acc[r];
    }
}

// ---------------------------------------------------------------------------
extern "C" void kernel_launch(void* const* d_in, const int* in_sizes, int n_in,
                              void* d_out, int out_size, void* d_ws, size_t ws_size,
                              hipStream_t stream)
{
    (void)in_sizes; (void)n_in; (void)out_size; (void)ws_size;
    const float* cam_code = (const float*)d_in[0];
    const float* proj_code = (const float*)d_in[1];
    const float* cam_w0  = (const float*)d_in[2];
    const float* cam_b0  = (const float*)d_in[3];
    const float* cam_w1  = (const float*)d_in[4];
    const float* cam_b1  = (const float*)d_in[5];
    const float* proj_w0 = (const float*)d_in[6];
    const float* proj_b0 = (const float*)d_in[7];
    const float* proj_w1 = (const float*)d_in[8];
    const float* proj_b1 = (const float*)d_in[9];
    float* out = (float*)d_out;

    // workspace layout (bytes): [0, 802816) packed weights, then packed B
    __bf16* wfrag = (__bf16*)d_ws;
    __bf16* bfrag = (__bf16*)((char*)d_ws + (size_t)8 * WFRAG_MAT * 2);

    // allow >64KB dynamic LDS (WGP has 320KB)
    (void)hipFuncSetAttribute((const void*)proj_kernel, hipFuncAttributeMaxDynamicSharedMemorySize, SMEM_BYTES);
    (void)hipFuncSetAttribute((const void*)cam_kernel,  hipFuncAttributeMaxDynamicSharedMemorySize, SMEM_BYTES);

    prep_wfrag<<<(8 * WFRAG_MAT) / 256, 256, 0, stream>>>(cam_w0, cam_w1, proj_w0, proj_w1, wfrag);
    proj_kernel<<<1024 / ROWS_WG, 256, SMEM_BYTES, stream>>>(proj_code, wfrag, proj_b0, proj_b1, bfrag);
    cam_kernel<<<(256 * 512) / ROWS_WG, 256, SMEM_BYTES, stream>>>(cam_code, wfrag, cam_b0, cam_b1, bfrag, out);
}
